// EdgeNetworkModel_37598143709436
// MI455X (gfx1250) — matrix operand order repaired and checked by reading, not compiled, not run on hardware
//
#include <hip/hip_runtime.h>
#include <hip/hip_bf16.h>

typedef __attribute__((ext_vector_type(16))) _Float16 v16h;
typedef __attribute__((ext_vector_type(8)))  _Float16 v8h;
typedef __attribute__((ext_vector_type(8)))  float    v8f;

// ---------------------------------------------------------------------------
// Fragment loaders (CDNA5 WMMA f16 layouts, wave32)
// A 16x32 f16: lane m=lane&15, g=lane>>4; halves 0..7 -> K = kb + g*8 + i,
//              halves 8..15 -> K = kb + 16 + g*8 + i   (row-major [16][kstride])
// B 32x16 f16 from W^T stored [64][K]: lane col=lane&15 (+nb*16), halves 0..7 ->
//              K = kb + g*8 + i, halves 8..15 -> K = kb + 16 + g*8 + i (same as A).
// C/D 16x16 f32: VGPR v -> M = g*8+v, N = lane&15 (+nb*16)
// ---------------------------------------------------------------------------
__device__ inline v16h load_a_frag(const _Float16* sA, int kstride, int m, int g, int kb) {
    const _Float16* p = sA + m * kstride + kb + g * 8;
    v8h lo = *(const v8h*)(p);
    v8h hi = *(const v8h*)(p + 16);
    return __builtin_shufflevector(lo, hi, 0,1,2,3,4,5,6,7,8,9,10,11,12,13,14,15);
}

__device__ inline v16h load_b_frag(const _Float16* wt, int col, int wks, int g, int kb) {
    const _Float16* p = wt + col * wks + kb + g * 8;
    v8h lo = *(const v8h*)(p);
    v8h hi = *(const v8h*)(p + 16);
    return __builtin_shufflevector(lo, hi, 0,1,2,3,4,5,6,7,8,9,10,11,12,13,14,15);
}

__device__ inline unsigned int pack2h(float a, float b) {
    union { _Float16 h[2]; unsigned int u; } u_;
    u_.h[0] = (_Float16)a; u_.h[1] = (_Float16)b;
    return u_.u;
}

// One GEMM stage: acc[4] (four 16-col blocks) over NK k-blocks of 32.
template<int NK>
__device__ inline void mm_stage(const _Float16* sA, int akstride, const int (&kbs)[NK],
                                const _Float16* wt, int wks,
                                int m, int g, int nlo, v8f acc[4]) {
#pragma unroll
    for (int nb = 0; nb < 4; ++nb)
#pragma unroll
        for (int v = 0; v < 8; ++v) acc[nb][v] = 0.0f;
#pragma unroll
    for (int s = 0; s < NK; ++s) {
        v16h a = load_a_frag(sA, akstride, m, g, kbs[s]);
#pragma unroll
        for (int nb = 0; nb < 4; ++nb) {
            v16h b = load_b_frag(wt, nb * 16 + nlo, wks, g, s * 32);
            acc[nb] = __builtin_amdgcn_wmma_f32_16x16x32_f16(
                false, a, false, b, (short)0, acc[nb], false, false);
        }
    }
}

// Bias (+optional ReLU) then store as f16 into row-major [16][192] LDS at colbase.
__device__ inline void store_act(_Float16* dst, int colbase, v8f acc[4],
                                 const float bias[4], bool do_relu, int g, int nlo) {
#pragma unroll
    for (int nb = 0; nb < 4; ++nb)
#pragma unroll
        for (int v = 0; v < 8; ++v) {
            float t = acc[nb][v] + bias[nb];
            if (do_relu) t = fmaxf(t, 0.0f);
            dst[(g * 8 + v) * 192 + colbase + nb * 16 + nlo] = (_Float16)t;
        }
}

// ---------------------------------------------------------------------------
// Weight transpose+convert: src [K][Nw] f32 -> dst [Nw][K] f16
// ---------------------------------------------------------------------------
__global__ void convT_kernel(const float* __restrict__ src, _Float16* __restrict__ dst,
                             int K, int Nw) {
    int tot = K * Nw;
    for (int i = blockIdx.x * blockDim.x + threadIdx.x; i < tot;
         i += gridDim.x * blockDim.x) {
        int k = i / Nw, n = i % Nw;
        dst[n * K + k] = (_Float16)src[i];
    }
}

__global__ void zero_f32(float* __restrict__ p, int n) {
    for (int i = blockIdx.x * blockDim.x + threadIdx.x; i < n;
         i += gridDim.x * blockDim.x) p[i] = 0.0f;
}

// ---------------------------------------------------------------------------
// Node projection: h = x @ pW + pb   (N x 128) @ (128 x 64), WMMA tiles of 16 rows
// ---------------------------------------------------------------------------
__global__ __launch_bounds__(128) void node_proj_kernel(
    const float* __restrict__ x, const _Float16* __restrict__ pWT,
    const float* __restrict__ pb, float* __restrict__ hbuf, int nnodes) {
    __shared__ __align__(32) _Float16 sPW[64 * 128];
    __shared__ __align__(32) _Float16 sX[4][16 * 128];

    const int tid = threadIdx.x;
    const int lane = tid & 31;
    const int wv = tid >> 5;
    {
        const v8h* src = (const v8h*)pWT;
        v8h* dst = (v8h*)sPW;
        for (int i = tid; i < (64 * 128) / 8; i += 128) dst[i] = src[i];
    }
    __syncthreads();

    const int m = lane & 15, g = lane >> 4, nlo = lane & 15;
    float bp[4];
#pragma unroll
    for (int nb = 0; nb < 4; ++nb) bp[nb] = pb[nb * 16 + nlo];

    const int ntiles = nnodes >> 4;
    for (int tile = blockIdx.x * 4 + wv; tile < ntiles; tile += gridDim.x * 4) {
        const size_t base = (size_t)tile * 16;
        // stage x tile (16x128 f32 -> f16)
        for (int i = lane; i < 512; i += 32) {
            float4 xv = *(const float4*)(x + base * 128 + (size_t)i * 4);
            _Float16* d = &sX[wv][i * 4];
            d[0] = (_Float16)xv.x; d[1] = (_Float16)xv.y;
            d[2] = (_Float16)xv.z; d[3] = (_Float16)xv.w;
        }
        v8f acc[4];
        constexpr int kbs[4] = {0, 32, 64, 96};
        mm_stage<4>(sX[wv], 128, kbs, sPW, 128, m, g, nlo, acc);
#pragma unroll
        for (int nb = 0; nb < 4; ++nb)
#pragma unroll
            for (int v = 0; v < 8; ++v)
                hbuf[(base + g * 8 + v) * 64 + nb * 16 + nlo] = acc[nb][v] + bp[nb];
    }
}

// ---------------------------------------------------------------------------
// Edge layer: full edge-MLP + message-MLP chain for 16 edges per wave,
// scatter-add into agg. 4 waves/block sharing one 56KB LDS weight image.
// ---------------------------------------------------------------------------
__global__ __launch_bounds__(128) void edge_layer_kernel(
    const float* __restrict__ hbuf,
    const int* __restrict__ eidx,            // [2][E] (row; col)
    const float* __restrict__ ea_f32,        // layer 0 input
    const _Float16* ea_f16_in,               // layers >0 input (aliases ea_out)
    _Float16* ea_out,
    float* __restrict__ agg,
    const _Float16* __restrict__ wts,        // [eW1T | eW2T | nW1T | nW2T] f16
    const float* __restrict__ eb1, const float* __restrict__ eb2,
    const float* __restrict__ nb1, const float* __restrict__ nb2,
    int nedges, int use_f32_ea) {
    __shared__ __align__(32) _Float16 sW[28672];         // 56 KB, shared by 4 waves
    __shared__ __align__(32) _Float16 sEinA[4][16 * 192]; // 4 x 6 KB
    __shared__ int sColsA[4][16];

    const int tid = threadIdx.x;
    const int lane = tid & 31;
    const int wv = tid >> 5;
    {
        const v8h* src = (const v8h*)wts;
        v8h* dst = (v8h*)sW;
        for (int i = tid; i < 28672 / 8; i += 128) dst[i] = src[i];
    }
    __syncthreads();   // single uniform barrier; none inside the tile loop

    _Float16* sEin = sEinA[wv];
    int* sCols = sColsA[wv];

    const int m = lane & 15, g = lane >> 4, nlo = lane & 15;
    float be1[4], be2[4], bn1[4], bn2[4];
#pragma unroll
    for (int nb = 0; nb < 4; ++nb) {
        be1[nb] = eb1[nb * 16 + nlo];
        be2[nb] = eb2[nb * 16 + nlo];
        bn1[nb] = nb1[nb * 16 + nlo];
        bn2[nb] = nb2[nb * 16 + nlo];
    }

    const _Float16* w_e1 = sW;             // kstride 192
    const _Float16* w_e2 = sW + 12288;     // kstride 64
    const _Float16* w_n1 = sW + 16384;     // kstride 128
    const _Float16* w_n2 = sW + 24576;     // kstride 64

    const int ntiles = nedges >> 4;
    for (int tile = blockIdx.x * 4 + wv; tile < ntiles; tile += gridDim.x * 4) {
        const int e0 = tile << 4;
        if (lane < 16) sCols[lane] = eidx[nedges + e0 + lane];
        // ---- gather: ein = [h[row] | h[col] | ea]  (16 x 192 f16 in LDS)
        // lanes 0..15 fetch h[row] (float4), lanes 16..31 fetch h[col] (float4)
        for (int e = 0; e < 16; ++e) {
            const int ge = e0 + e;
            const int r = eidx[ge];
            const int c = eidx[nedges + ge];
            const float* src = hbuf + (size_t)(g ? c : r) * 64 + nlo * 4;
            float4 hv = *(const float4*)src;
            _Float16* d = &sEin[e * 192 + g * 64 + nlo * 4];
            d[0] = (_Float16)hv.x; d[1] = (_Float16)hv.y;
            d[2] = (_Float16)hv.z; d[3] = (_Float16)hv.w;
            if (use_f32_ea) {
                float2 ev = *(const float2*)(ea_f32 + (size_t)ge * 64 + lane * 2);
                *(unsigned int*)&sEin[e * 192 + 128 + lane * 2] = pack2h(ev.x, ev.y);
            } else {
                *(unsigned int*)&sEin[e * 192 + 128 + lane * 2] =
                    *(const unsigned int*)(ea_f16_in + (size_t)ge * 64 + lane * 2);
            }
        }
        // (intra-wave: LDS ops are in-order, no barrier needed)
        v8f acc[4];
        // ---- t1 = relu(ein @ eW1 + eb1) : K=192, out -> cols 64..127
        constexpr int kb1[6] = {0, 32, 64, 96, 128, 160};
        mm_stage<6>(sEin, 192, kb1, w_e1, 192, m, g, nlo, acc);
        store_act(sEin, 64, acc, be1, true, g, nlo);
        // ---- ea' = t1 @ eW2 + eb2 : K=64, out -> cols 128..191 + global
        constexpr int kb2[2] = {64, 96};
        mm_stage<2>(sEin, 192, kb2, w_e2, 64, m, g, nlo, acc);
#pragma unroll
        for (int nb = 0; nb < 4; ++nb)
#pragma unroll
            for (int v = 0; v < 8; ++v) {
                const int M = g * 8 + v;
                _Float16 th = (_Float16)(acc[nb][v] + be2[nb]);
                sEin[M * 192 + 128 + nb * 16 + nlo] = th;
                ea_out[(size_t)(e0 + M) * 64 + nb * 16 + nlo] = th;
            }
        // ---- t2 = relu([h_row | ea'] @ nW1 + nb1) : K=128 (cols 0..63,128..191)
        constexpr int kb3[4] = {0, 32, 128, 160};
        mm_stage<4>(sEin, 192, kb3, w_n1, 128, m, g, nlo, acc);
        store_act(sEin, 64, acc, bn1, true, g, nlo);
        // ---- msg = t2 @ nW2 + nb2 : K=64, scatter-add into agg[col]
        constexpr int kb4[2] = {64, 96};
        mm_stage<2>(sEin, 192, kb4, w_n2, 64, m, g, nlo, acc);
#pragma unroll
        for (int nb = 0; nb < 4; ++nb)
#pragma unroll
            for (int v = 0; v < 8; ++v) {
                const int M = g * 8 + v;
                atomicAdd(&agg[(size_t)sCols[M] * 64 + nb * 16 + nlo],
                          acc[nb][v] + bn2[nb]);
            }
    }
}

// ---------------------------------------------------------------------------
// h = relu(agg + h); agg = 0 (ready for next layer)
// ---------------------------------------------------------------------------
__global__ void node_update_kernel(float* __restrict__ h, float* __restrict__ agg, int n) {
    for (int i = blockIdx.x * blockDim.x + threadIdx.x; i < n;
         i += gridDim.x * blockDim.x) {
        float v = agg[i] + h[i];
        h[i] = fmaxf(v, 0.0f);
        agg[i] = 0.0f;
    }
}

// ---------------------------------------------------------------------------
// Mean pool (sum into pooled[64] via per-block partial reduce + atomics)
// ---------------------------------------------------------------------------
__global__ __launch_bounds__(256) void pool_kernel(const float* __restrict__ h,
                                                   float* __restrict__ pooled, int nn) {
    __shared__ float red[256];
    const int t = threadIdx.x, f = t & 63, rg = t >> 6;
    float local = 0.0f;
    for (int r = blockIdx.x * 4 + rg; r < nn; r += gridDim.x * 4)
        local += h[(size_t)r * 64 + f];
    red[t] = local;
    __syncthreads();
    if (t < 64)
        atomicAdd(&pooled[f], red[t] + red[t + 64] + red[t + 128] + red[t + 192]);
}

__global__ void final_kernel(const float* __restrict__ pooled, const float* __restrict__ lW,
                             const float* __restrict__ lb, float* __restrict__ out, int nn) {
    __shared__ float sp[64];
    const int t = threadIdx.x;
    sp[t] = pooled[t] / (float)nn;
    __syncthreads();
    if (t < 3) {
        float s = lb[t];
        for (int f = 0; f < 64; ++f) s += sp[f] * lW[f * 3 + t];
        out[t] = s;
    }
}

// ---------------------------------------------------------------------------
extern "C" void kernel_launch(void* const* d_in, const int* in_sizes, int n_in,
                              void* d_out, int out_size, void* d_ws, size_t ws_size,
                              hipStream_t stream) {
    const float* x     = (const float*)d_in[0];
    const int*   eidx  = (const int*)d_in[1];
    const float* eattr = (const float*)d_in[2];
    const float* pW  = (const float*)d_in[3];  const float* pb  = (const float*)d_in[4];
    const float* eW1 = (const float*)d_in[5];  const float* eb1 = (const float*)d_in[6];
    const float* eW2 = (const float*)d_in[7];  const float* eb2 = (const float*)d_in[8];
    const float* nW1 = (const float*)d_in[9];  const float* nb1 = (const float*)d_in[10];
    const float* nW2 = (const float*)d_in[11]; const float* nb2 = (const float*)d_in[12];
    const float* lW  = (const float*)d_in[13]; const float* lb  = (const float*)d_in[14];
    float* out = (float*)d_out;

    const int Nn = in_sizes[0] / 128;
    const int Ee = in_sizes[1] / 2;

    char* ws = (char*)d_ws;
    size_t off = 0;
    auto walloc = [&](size_t bytes) -> void* {
        void* p = ws + off;
        off = (off + bytes + 255) & ~(size_t)255;
        return p;
    };
    float*    hbuf  = (float*)walloc((size_t)Nn * 64 * 4);
    float*    agg   = (float*)walloc((size_t)Nn * 64 * 4);
    _Float16* ea    = (_Float16*)walloc((size_t)Ee * 64 * 2);
    _Float16* pWT   = (_Float16*)walloc((size_t)64 * 128 * 2);
    _Float16* wts   = (_Float16*)walloc((size_t)3 * 28672 * 2);
    float*    pooled= (float*)walloc(64 * 4);
    (void)ws_size; (void)n_in; (void)out_size;

    zero_f32<<<512, 256, 0, stream>>>(agg, Nn * 64);
    zero_f32<<<1, 64, 0, stream>>>(pooled, 64);

    convT_kernel<<<32, 256, 0, stream>>>(pW, pWT, 128, 64);
    for (int l = 0; l < 3; ++l) {
        _Float16* wl = wts + (size_t)l * 28672;
        convT_kernel<<<48, 256, 0, stream>>>(eW1 + (size_t)l * 192 * 64, wl,          192, 64);
        convT_kernel<<<16, 256, 0, stream>>>(eW2 + (size_t)l * 64 * 64,  wl + 12288,  64, 64);
        convT_kernel<<<32, 256, 0, stream>>>(nW1 + (size_t)l * 128 * 64, wl + 16384, 128, 64);
        convT_kernel<<<16, 256, 0, stream>>>(nW2 + (size_t)l * 64 * 64,  wl + 24576,  64, 64);
    }

    node_proj_kernel<<<1024, 128, 0, stream>>>(x, pWT, pb, hbuf, Nn);

    for (int l = 0; l < 3; ++l) {
        edge_layer_kernel<<<2048, 128, 0, stream>>>(
            hbuf, eidx, eattr, ea, ea, agg, wts + (size_t)l * 28672,
            eb1 + l * 64, eb2 + l * 64, nb1 + l * 64, nb2 + l * 64,
            Ee, (l == 0) ? 1 : 0);
        node_update_kernel<<<1024, 256, 0, stream>>>(hbuf, agg, Nn * 64);
    }

    pool_kernel<<<512, 256, 0, stream>>>(hbuf, pooled, Nn);
    final_kernel<<<1, 64, 0, stream>>>(pooled, lW, lb, out, Nn);
}
